// OptHIST_20014547599815
// MI455X (gfx1250) — compile-verified, matrix operand order checked by reading, not run on hardware
//
#include <hip/hip_runtime.h>

// ---------------------------------------------------------------------------
// HIST model on gfx1250: fused 2-layer GRU + flash-attention concept phases.
// GEMMs: v_wmma_f32_16x16x32_bf16 (wave32). Tile staging: Tensor Data Mover.
// ---------------------------------------------------------------------------

#define N_STOCKS   8192
#define N_CONCEPTS 4096
#define HID        128
#define SEQ_T      64
#define IN_D       6

typedef unsigned short us;
typedef __bf16        bf16x16 __attribute__((ext_vector_type(16)));
typedef float         f32x8   __attribute__((ext_vector_type(8)));
typedef unsigned int  u32x4   __attribute__((ext_vector_type(4)));
typedef int           i32x4v  __attribute__((ext_vector_type(4)));
typedef int           i32x8v  __attribute__((ext_vector_type(8)));

__device__ __forceinline__ us f2bf_us(float f) {
    union { float f; unsigned u; } v; v.f = f;
    unsigned r = (v.u + 0x7FFFu + ((v.u >> 16) & 1u)) >> 16;   // RNE
    return (us)r;
}
__device__ __forceinline__ float bf2f(us s) {
    union { unsigned u; float f; } v; v.u = ((unsigned)s) << 16;
    return v.f;
}
__device__ __forceinline__ float sigm(float x) { return 1.f / (1.f + __expf(-x)); }
__device__ __forceinline__ f32x8 zero8() {
    f32x8 z;
#pragma unroll
    for (int i = 0; i < 8; ++i) z[i] = 0.f;
    return z;
}

struct Pack32 { u32x4 lo, hi; };

// A-frag (16x32, row-major MxK) / B-frag (stored [n][k], i.e. B^T row-major)
// lane<16: row=lane, k=kBase..+15 ; lane>=16: row=lane-16, k=kBase+16..+31
// 16B-aligned vector loads (ld multiple of 8 elems, kBase multiple of 16).
__device__ __forceinline__ bf16x16 frag_ld_lds(const us* buf, int ld, int rowBase, int kBase) {
    const int lane = threadIdx.x & 31;
    const us* p = buf + (rowBase + (lane & 15)) * ld + kBase + ((lane >> 4) << 4);
    Pack32 pk;
    pk.lo = *(const u32x4*)(p);
    pk.hi = *(const u32x4*)(p + 8);
    return __builtin_bit_cast(bf16x16, pk);
}
__device__ __forceinline__ bf16x16 frag_ld_gf(const us* buf, int ld, int rowBase, int kBase) {
    const int lane = threadIdx.x & 31;
    const us* p = buf + (rowBase + (lane & 15)) * ld + kBase + ((lane >> 4) << 4);
    Pack32 pk;
    pk.lo = *(const u32x4*)(p);
    pk.hi = *(const u32x4*)(p + 8);
    return __builtin_bit_cast(bf16x16, pk);
}
__device__ __forceinline__ f32x8 wmma_bf16(bf16x16 a, bf16x16 b, f32x8 c) {
    return __builtin_amdgcn_wmma_f32_16x16x32_bf16(false, a, false, b, (short)0, c, false, false);
}

// ---------------------------------------------------------------------------
// Tensor Data Mover: 1-D tile of `nelems` bf16 from global -> LDS byte offset.
// D# per CDNA5 ISA 8.3-8.5: group0 {count=1, lds_addr, global_addr, type=2},
// group1 {data_size=2B, tensor_dim0=tile_dim0=nelems, dims1=1, stride0=nelems}.
__device__ __forceinline__ void tdm_load_2b(unsigned ldsOff, const void* gp, unsigned nelems) {
    const unsigned long long ga = (unsigned long long)gp;
    u32x4 g0 = { 1u,                                   // count=1 (valid user D#)
                 ldsOff,                               // lds_addr (bytes)
                 (unsigned)(ga & 0xFFFFFFFFu),         // global_addr[31:0]
                 ((unsigned)((ga >> 32) & 0x01FFFFFFu)) | 0x80000000u }; // [56:32] | type=2
    i32x8v g1 = { (int)0x00010000,                     // data_size=1 (2 bytes/elem)
                  (int)((nelems & 0xFFFFu) << 16),     // tensor_dim0[15:0] @ bits 63:48
                  (int)(((nelems >> 16) & 0xFFFFu) | (1u << 16)), // dim0 hi | tensor_dim1=1
                  (int)((nelems & 0xFFFFu) << 16),     // tile_dim0 @ bits 127:112
                  1,                                   // tile_dim1 = 1
                  (int)nelems,                         // tensor_dim0_stride[31:0]
                  0, 0 };
    i32x4v gz = { 0, 0, 0, 0 };
#if __clang_major__ >= 23
    i32x8v gz8 = { 0, 0, 0, 0, 0, 0, 0, 0 };
    __builtin_amdgcn_tensor_load_to_lds(g0, g1, gz, gz, gz8, 0);
#else
    __builtin_amdgcn_tensor_load_to_lds(g0, g1, gz, gz, 0);
#endif
}
__device__ __forceinline__ void tdm_wait() {
    __builtin_amdgcn_s_wait_tensorcnt(0);
}

// ---------------------------------------------------------------------------
// elementwise f32 -> bf16
__global__ void k_cvt(const float* __restrict__ in, us* __restrict__ out, int n) {
    int i = blockIdx.x * blockDim.x + threadIdx.x;
    if (i < n) out[i] = f2bf_us(in[i]);
}
// f32 (rows x inLd) -> bf16 (rows x outLd) zero-padded rows
__global__ void k_cvt_pad(const float* __restrict__ in, us* __restrict__ out,
                          int rows, int inLd, int outLd) {
    int e = blockIdx.x * blockDim.x + threadIdx.x;
    if (e >= rows * outLd) return;
    int r = e / outLd, c = e - r * outLd;
    out[e] = (c < inLd) ? f2bf_us(in[r * inLd + c]) : (us)0;
}

// colsum[c] = sum_i cm[i,c]*mv[i]
__global__ void k_colsum(const float* __restrict__ cm, const float* __restrict__ mv,
                         float* __restrict__ colsum, int n, int c_dim) {
    int c = blockIdx.x * blockDim.x + threadIdx.x;
    if (c >= c_dim) return;
    float s = 0.f;
    for (int i = 0; i < n; ++i) s += cm[(size_t)i * c_dim + c] * mv[i];
    colsum[c] = s;
}

// per-row L2 norm: normalized bf16, optional raw bf16, optional diag=(||x||/(||x||+eps))^2
__global__ void k_rownorm(const float* __restrict__ X, int rows,
                          us* __restrict__ outN, us* __restrict__ outRaw,
                          float* __restrict__ diagOut) {
    const int wave = threadIdx.x >> 5, lane = threadIdx.x & 31;
    const int row = blockIdx.x * 8 + wave;
    if (row >= rows) return;
    const float* x = X + (size_t)row * HID;
    float v[4], s = 0.f;
#pragma unroll
    for (int i = 0; i < 4; ++i) { v[i] = x[lane + 32 * i]; s += v[i] * v[i]; }
#pragma unroll
    for (int off = 16; off; off >>= 1) s += __shfl_xor(s, off, 32);
    const float nrm = sqrtf(s);
    const float inv = 1.f / (nrm + 1e-12f);
#pragma unroll
    for (int i = 0; i < 4; ++i) {
        const int c = lane + 32 * i;
        outN[(size_t)row * HID + c] = f2bf_us(v[i] * inv);
        if (outRaw) outRaw[(size_t)row * HID + c] = f2bf_us(v[i]);
    }
    if (diagOut && lane == 0) { float d = nrm * inv; diagOut[row] = d * d; }
}

// ---------------------------------------------------------------------------
// Fused 2-layer GRU over T=64 steps; 64 stocks per workgroup, states in LDS,
// weights (bf16, L2-resident) streamed as aligned b128 fragments. LDS: 164 KB.
// Gate routing uses the scalar tile index nt (uniform) -> scalar branch only.
__global__ void k_gru(const float* __restrict__ xg,
                      const us* __restrict__ wih0, const us* __restrict__ whh0,
                      const float* __restrict__ bih0, const float* __restrict__ bhh0,
                      const us* __restrict__ wih1, const us* __restrict__ whh1,
                      const float* __restrict__ bih1, const float* __restrict__ bhh1,
                      float* __restrict__ x0out) {
    extern __shared__ char smem[];
    us*    sH1 = (us*)smem;                 // 64x128
    us*    sH2 = sH1 + 64 * 128;            // 64x128
    us*    sX  = sH2 + 64 * 128;            // 64x32 (zero-padded K)
    float* gRZ = (float*)(sX + 64 * 32);    // 64x256 (r|z)
    float* gIN = gRZ + 64 * 256;            // 64x128
    float* gHN = gIN + 64 * 128;            // 64x128
    const int r0 = blockIdx.x * 64;
    const int wave = threadIdx.x >> 5, lane = threadIdx.x & 31;

    for (int e = threadIdx.x; e < 64 * 128; e += 256) { sH1[e] = 0; sH2[e] = 0; }
    __syncthreads();

    for (int t = 0; t < SEQ_T; ++t) {
        // ---- layer 0: stage x_t (seq[m,t,d] = x[m, d*64 + t]), pad K to 32
        for (int e = threadIdx.x; e < 64 * 32; e += 256) {
            const int m = e >> 5, d = e & 31;
            const float v = (d < IN_D) ? xg[(size_t)(r0 + m) * (IN_D * SEQ_T) + d * SEQ_T + t] : 0.f;
            sX[e] = f2bf_us(v);
        }
        __syncthreads();
        // gh = h1 @ whh0^T
        for (int tt = wave; tt < 96; tt += 8) {
            const int mt = tt & 3, nt = tt >> 2;
            f32x8 a = zero8();
#pragma unroll
            for (int kk = 0; kk < 4; ++kk)
                a = wmma_bf16(frag_ld_lds(sH1, 128, mt * 16, kk * 32),
                              frag_ld_gf(whh0, 128, nt * 16, kk * 32), a);
            const int n = nt * 16 + (lane & 15), mB = mt * 16 + ((lane >> 4) << 3);
            if (nt < 16) {                      // r|z gates: uniform scalar branch
#pragma unroll
                for (int r = 0; r < 8; ++r) gRZ[(mB + r) * 256 + n] = a[r];
            } else {                            // n-gate (h part)
#pragma unroll
                for (int r = 0; r < 8; ++r) gHN[(mB + r) * 128 + (n - 256)] = a[r];
            }
        }
        __syncthreads();
        // gi = x_t @ wih0^T  (wih0 padded to 384x32, zeros beyond K=6)
        for (int tt = wave; tt < 96; tt += 8) {
            const int mt = tt & 3, nt = tt >> 2;
            f32x8 a = wmma_bf16(frag_ld_lds(sX, 32, mt * 16, 0),
                                frag_ld_gf(wih0, 32, nt * 16, 0), zero8());
            const int n = nt * 16 + (lane & 15), mB = mt * 16 + ((lane >> 4) << 3);
            if (nt < 16) {
#pragma unroll
                for (int r = 0; r < 8; ++r) gRZ[(mB + r) * 256 + n] += a[r] + bih0[n] + bhh0[n];
            } else {
#pragma unroll
                for (int r = 0; r < 8; ++r) gIN[(mB + r) * 128 + (n - 256)] = a[r] + bih0[n];
            }
        }
        __syncthreads();
        // cell update h1
        for (int e = threadIdx.x; e < 64 * 128; e += 256) {
            const int m = e >> 7, h = e & 127;
            const float rr = sigm(gRZ[m * 256 + h]);
            const float zz = sigm(gRZ[m * 256 + 128 + h]);
            const float nn = tanhf(gIN[e] + rr * (gHN[e] + bhh0[256 + h]));
            sH1[e] = f2bf_us((1.f - zz) * nn + zz * bf2f(sH1[e]));
        }
        __syncthreads();
        // ---- layer 1: input sH1, state sH2
        for (int tt = wave; tt < 96; tt += 8) {
            const int mt = tt & 3, nt = tt >> 2;
            f32x8 a = zero8();
#pragma unroll
            for (int kk = 0; kk < 4; ++kk)
                a = wmma_bf16(frag_ld_lds(sH2, 128, mt * 16, kk * 32),
                              frag_ld_gf(whh1, 128, nt * 16, kk * 32), a);
            const int n = nt * 16 + (lane & 15), mB = mt * 16 + ((lane >> 4) << 3);
            if (nt < 16) {
#pragma unroll
                for (int r = 0; r < 8; ++r) gRZ[(mB + r) * 256 + n] = a[r];
            } else {
#pragma unroll
                for (int r = 0; r < 8; ++r) gHN[(mB + r) * 128 + (n - 256)] = a[r];
            }
        }
        __syncthreads();
        for (int tt = wave; tt < 96; tt += 8) {
            const int mt = tt & 3, nt = tt >> 2;
            f32x8 a = zero8();
#pragma unroll
            for (int kk = 0; kk < 4; ++kk)
                a = wmma_bf16(frag_ld_lds(sH1, 128, mt * 16, kk * 32),
                              frag_ld_gf(wih1, 128, nt * 16, kk * 32), a);
            const int n = nt * 16 + (lane & 15), mB = mt * 16 + ((lane >> 4) << 3);
            if (nt < 16) {
#pragma unroll
                for (int r = 0; r < 8; ++r) gRZ[(mB + r) * 256 + n] += a[r] + bih1[n] + bhh1[n];
            } else {
#pragma unroll
                for (int r = 0; r < 8; ++r) gIN[(mB + r) * 128 + (n - 256)] = a[r] + bih1[n];
            }
        }
        __syncthreads();
        for (int e = threadIdx.x; e < 64 * 128; e += 256) {
            const int m = e >> 7, h = e & 127;
            const float rr = sigm(gRZ[m * 256 + h]);
            const float zz = sigm(gRZ[m * 256 + 128 + h]);
            const float nn = tanhf(gIN[e] + rr * (gHN[e] + bhh1[256 + h]));
            sH2[e] = f2bf_us((1.f - zz) * nn + zz * bf2f(sH2[e]));
        }
        __syncthreads();
    }
    for (int e = threadIdx.x; e < 64 * 128; e += 256)
        x0out[(size_t)r0 * 128 + e] = bf2f(sH2[e]);
}

// ---------------------------------------------------------------------------
// initial_rep: (C x H) = W^T @ x0 with on-the-fly A-weights from concept matrix
__global__ void k_irep(const float* __restrict__ cm, const float* __restrict__ mv,
                       const float* __restrict__ colsum, const us* __restrict__ x0r,
                       float* __restrict__ irep) {
    extern __shared__ char smem[];
    us* sA = (us*)smem;          // 128(c) x 32(i)
    us* sB = sA + 128 * 32;      // 128(h) x 32(i)  (x0 transposed)
    const int c0 = blockIdx.x * 128;
    const int wave = threadIdx.x >> 5, lane = threadIdx.x & 31;
    f32x8 acc[8];
#pragma unroll
    for (int nt = 0; nt < 8; ++nt) acc[nt] = zero8();

    for (int i0 = 0; i0 < N_STOCKS; i0 += 32) {
        for (int e = threadIdx.x; e < 128 * 32; e += 256) {
            const int i = e >> 7, c = e & 127;
            float w = cm[(size_t)(i0 + i) * N_CONCEPTS + (c0 + c)];
            w = (w * mv[i0 + i]) / (w * colsum[c0 + c] + 1.0f);
            sA[c * 32 + i] = f2bf_us(w);
        }
        for (int e = threadIdx.x; e < 128 * 32; e += 256) {
            const int i = e >> 7, h = e & 127;
            sB[h * 32 + i] = x0r[(size_t)(i0 + i) * HID + h];
        }
        __syncthreads();
#pragma unroll
        for (int nt = 0; nt < 8; ++nt)
            acc[nt] = wmma_bf16(frag_ld_lds(sA, 32, wave * 16, 0),
                                frag_ld_lds(sB, 32, nt * 16, 0), acc[nt]);
        __syncthreads();
    }
    for (int nt = 0; nt < 8; ++nt) {
        const int n = nt * 16 + (lane & 15), mB = wave * 16 + ((lane >> 4) << 3);
#pragma unroll
        for (int r = 0; r < 8; ++r)
            irep[(size_t)(c0 + mB + r) * HID + n] = acc[nt][r];
    }
}

// ---------------------------------------------------------------------------
// Flash attention: O_i = sum_j softmax_j(Qn_i . Kn_j) * V_j  (H=128, 128 Q rows/block)
// Q and K tiles staged by the Tensor Data Mover (wave 0), V transposed by VALU.
__global__ void k_attn(const us* __restrict__ Qn, const us* __restrict__ Kn,
                       const us* __restrict__ Vb, float* __restrict__ O, int L) {
    extern __shared__ char smem[];
    us*    sQ  = (us*)smem;                    // 128x128   (LDS off 0)
    us*    sK  = sQ + 128 * 128;               // 128x128   (LDS off 32768)
    us*    sVt = sK + 128 * 128;               // 128(h) x 128(j)
    float* sS  = (float*)(sVt + 128 * 128);    // 128x128
    us*    sP  = (us*)(sS + 128 * 128);        // 128x128
    float* sMax   = (float*)(sP + 128 * 128);  // 128
    float* sSum   = sMax + 128;
    float* sAlpha = sSum + 128;

    const int r0 = blockIdx.x * 128;
    const int wave = threadIdx.x >> 5, lane = threadIdx.x & 31;
    if (wave == 0) {
        tdm_load_2b(0u, Qn + (size_t)r0 * 128, 128 * 128);   // async Q tile
    }
    if (threadIdx.x < 128) { sMax[threadIdx.x] = -3.0e38f; sSum[threadIdx.x] = 0.f; }
    f32x8 acc[8];
#pragma unroll
    for (int nt = 0; nt < 8; ++nt) acc[nt] = zero8();
    if (wave == 0) tdm_wait();
    __syncthreads();

    for (int j0 = 0; j0 < L; j0 += 128) {
        if (wave == 0) {
            tdm_load_2b(128 * 128 * 2, Kn + (size_t)j0 * 128, 128 * 128);  // async K tile
        }
        for (int e = threadIdx.x; e < 128 * 128; e += 256) {
            const int j = e >> 7, h = e & 127;
            sVt[h * 128 + j] = Vb[(size_t)(j0 + j) * 128 + h];
        }
        if (wave == 0) tdm_wait();
        __syncthreads();
        // S = Qn @ Kn^T, wave owns rows [16w,16w+16)
        for (int nt = 0; nt < 8; ++nt) {
            f32x8 s = zero8();
#pragma unroll
            for (int kk = 0; kk < 4; ++kk)
                s = wmma_bf16(frag_ld_lds(sQ, 128, wave * 16, kk * 32),
                              frag_ld_lds(sK, 128, nt * 16, kk * 32), s);
            const int n = nt * 16 + (lane & 15), mB = wave * 16 + ((lane >> 4) << 3);
#pragma unroll
            for (int r = 0; r < 8; ++r) sS[(mB + r) * 128 + n] = s[r];
        }
        __syncthreads();
        // online softmax per row
        if (threadIdx.x < 128) {
            const int i = threadIdx.x;
            float* row = sS + i * 128;
            float tmax = -3.0e38f;
            for (int j = 0; j < 128; ++j) tmax = fmaxf(tmax, row[j]);
            const float mOld = sMax[i];
            const float mNew = fmaxf(mOld, tmax);
            const float alpha = __expf(mOld - mNew);
            float psum = 0.f;
            for (int j = 0; j < 128; ++j) {
                const float p = __expf(row[j] - mNew);
                psum += p;
                sP[i * 128 + j] = f2bf_us(p);
            }
            sSum[i] = sSum[i] * alpha + psum;
            sMax[i] = mNew;
            sAlpha[i] = alpha;
        }
        __syncthreads();
        // acc = acc*alpha + P @ V
        for (int nt = 0; nt < 8; ++nt) {
            const int mB = wave * 16 + ((lane >> 4) << 3);
#pragma unroll
            for (int r = 0; r < 8; ++r) acc[nt][r] *= sAlpha[mB + r];
#pragma unroll
            for (int kk = 0; kk < 4; ++kk)
                acc[nt] = wmma_bf16(frag_ld_lds(sP, 128, wave * 16, kk * 32),
                                    frag_ld_lds(sVt, 128, nt * 16, kk * 32), acc[nt]);
        }
        __syncthreads();
    }
    for (int nt = 0; nt < 8; ++nt) {
        const int n = nt * 16 + (lane & 15), mB = wave * 16 + ((lane >> 4) << 3);
#pragma unroll
        for (int r = 0; r < 8; ++r) {
            const int m = mB + r;
            const float l = sSum[m];
            O[(size_t)(r0 + m) * 128 + n] = acc[nt][r] / (l > 0.f ? l : 1.f);
        }
    }
}

// ---------------------------------------------------------------------------
// out = [sub -] act(X @ W^T + b). X: f32 (converted while staging), W: bf16 via TDM.
__global__ void k_fc(const float* __restrict__ X, const us* __restrict__ Wb,
                     const float* __restrict__ bias, float* __restrict__ out,
                     const float* __restrict__ sub, int actLrelu) {
    extern __shared__ char smem[];
    us* sX = (us*)smem;          // 128x128  (LDS off 0)
    us* sW = sX + 128 * 128;     // 128x128  (LDS off 32768)
    const int r0 = blockIdx.x * 128;
    const int wave = threadIdx.x >> 5, lane = threadIdx.x & 31;
    if (wave == 0) tdm_load_2b(128 * 128 * 2, Wb, 128 * 128);   // async W tile
    for (int e = threadIdx.x; e < 128 * 128; e += 256) sX[e] = f2bf_us(X[(size_t)r0 * 128 + e]);
    if (wave == 0) tdm_wait();
    __syncthreads();
    for (int nt = 0; nt < 8; ++nt) {
        f32x8 acc = zero8();
#pragma unroll
        for (int kk = 0; kk < 4; ++kk)
            acc = wmma_bf16(frag_ld_lds(sX, 128, wave * 16, kk * 32),
                            frag_ld_lds(sW, 128, nt * 16, kk * 32), acc);
        const int n = nt * 16 + (lane & 15), mB = wave * 16 + ((lane >> 4) << 3);
#pragma unroll
        for (int r = 0; r < 8; ++r) {
            const int m = mB + r;
            float v = acc[r] + bias[n];
            if (actLrelu) v = (v >= 0.f) ? v : 0.01f * v;
            if (sub) v = sub[(size_t)(r0 + m) * 128 + n] - v;
            out[(size_t)(r0 + m) * 128 + n] = v;
        }
    }
}

// ---------------------------------------------------------------------------
// Top-3 of cos(x1,x1) per row (diag treated as 0), WMMA tiles, no N^2 buffer.
__global__ void k_topk(const us* __restrict__ Xn, float* __restrict__ topv, int* __restrict__ topi) {
    extern __shared__ char smem[];
    us*    sQ = (us*)smem;                  // 128x128  (LDS off 0)
    us*    sK = sQ + 128 * 128;             // 128x128  (LDS off 32768)
    float* sS = (float*)(sK + 128 * 128);   // 128x128
    const int r0 = blockIdx.x * 128;
    const int wave = threadIdx.x >> 5, lane = threadIdx.x & 31;
    if (wave == 0) {
        tdm_load_2b(0u, Xn + (size_t)r0 * 128, 128 * 128);
        tdm_wait();
    }
    float v0 = -3e38f, v1 = -3e38f, v2 = -3e38f;
    int   i0 = 0, i1 = 0, i2 = 0;
    __syncthreads();

    for (int j0 = 0; j0 < N_STOCKS; j0 += 128) {
        if (wave == 0) {
            tdm_load_2b(128 * 128 * 2, Xn + (size_t)j0 * 128, 128 * 128);
            tdm_wait();
        }
        __syncthreads();
        for (int nt = 0; nt < 8; ++nt) {
            f32x8 s = zero8();
#pragma unroll
            for (int kk = 0; kk < 4; ++kk)
                s = wmma_bf16(frag_ld_lds(sQ, 128, wave * 16, kk * 32),
                              frag_ld_lds(sK, 128, nt * 16, kk * 32), s);
            const int n = nt * 16 + (lane & 15), mB = wave * 16 + ((lane >> 4) << 3);
#pragma unroll
            for (int r = 0; r < 8; ++r) sS[(mB + r) * 128 + n] = s[r];
        }
        __syncthreads();
        if (threadIdx.x < 128) {
            const int i = threadIdx.x;
            const float* row = sS + i * 128;
            for (int j = 0; j < 128; ++j) {
                const int jg = j0 + j;
                float s = row[j];
                if (jg == r0 + i) s = 0.f;          // diag zeroed before top-k
                if (s > v0)      { v2 = v1; i2 = i1; v1 = v0; i1 = i0; v0 = s; i0 = jg; }
                else if (s > v1) { v2 = v1; i2 = i1; v1 = s;  i1 = jg; }
                else if (s > v2) { v2 = s;  i2 = jg; }
            }
        }
        __syncthreads();
    }
    if (threadIdx.x < 128) {
        const int i = r0 + threadIdx.x;
        topv[i * 3 + 0] = v0; topi[i * 3 + 0] = i0;
        topv[i * 3 + 1] = v1; topi[i * 3 + 1] = i1;
        topv[i * 3 + 2] = v2; topi[i * 3 + 2] = i2;
    }
}

// hidden_rep[j] += val * x1[i] for each top-k pair; one wave per (i,k)
__global__ void k_scatter(const float* __restrict__ topv, const int* __restrict__ topi,
                          const float* __restrict__ X, float* __restrict__ hrep,
                          int* __restrict__ flags, int total) {
    const int wave = threadIdx.x >> 5, lane = threadIdx.x & 31;
    const int p = blockIdx.x * 8 + wave;
    if (p >= total) return;
    const float v = topv[p];
    const int   j = topi[p];
    const int   i = p / 3;
    if (v == 0.f) return;
#pragma unroll
    for (int c = lane; c < HID; c += 32)
        atomicAdd(&hrep[(size_t)j * HID + c], v * X[(size_t)i * HID + c]);
    if (lane == 0) atomicOr(&flags[j], 1);
}

// diagonal contribution: hrep[j] += diag[j]*x1[j] where column j was selected
__global__ void k_diag(const int* __restrict__ flags, const float* __restrict__ diagv,
                       const float* __restrict__ X, float* __restrict__ hrep, int n) {
    const int idx = blockIdx.x * blockDim.x + threadIdx.x;
    if (idx >= n) return;
    const int j = idx >> 7;
    if (flags[j]) hrep[idx] += diagv[j] * X[idx];
}

// pred[i] = b_out + sum_h (pf+hf+ind)[i,h]*W_out[h]
__global__ void k_final(const float* __restrict__ pf, const float* __restrict__ hf,
                        const float* __restrict__ ind, const float* __restrict__ Wout,
                        const float* __restrict__ bout, float* __restrict__ pred, int n) {
    const int wave = threadIdx.x >> 5, lane = threadIdx.x & 31;
    const int i = blockIdx.x * 8 + wave;
    if (i >= n) return;
    float s = 0.f;
#pragma unroll
    for (int c = lane; c < HID; c += 32) {
        const size_t o = (size_t)i * HID + c;
        s += (pf[o] + hf[o] + ind[o]) * Wout[c];
    }
#pragma unroll
    for (int off = 16; off; off >>= 1) s += __shfl_xor(s, off, 32);
    if (lane == 0) pred[i] = s + bout[0];
}

// ---------------------------------------------------------------------------
extern "C" void kernel_launch(void* const* d_in, const int* in_sizes, int n_in,
                              void* d_out, int out_size, void* d_ws, size_t ws_size,
                              hipStream_t stream) {
    (void)in_sizes; (void)n_in; (void)out_size; (void)ws_size;
    const float* X     = (const float*)d_in[0];
    const float* CM    = (const float*)d_in[1];
    const float* MV    = (const float*)d_in[2];
    const float* wih0f = (const float*)d_in[3];
    const float* whh0f = (const float*)d_in[4];
    const float* bih0  = (const float*)d_in[5];
    const float* bhh0  = (const float*)d_in[6];
    const float* wih1f = (const float*)d_in[7];
    const float* whh1f = (const float*)d_in[8];
    const float* bih1  = (const float*)d_in[9];
    const float* bhh1  = (const float*)d_in[10];
    const float* W_ps  = (const float*)d_in[11];
    const float* b_ps  = (const float*)d_in[12];
    const float* W_psf = (const float*)d_in[13];
    const float* b_psf = (const float*)d_in[14];
    const float* W_psb = (const float*)d_in[15];
    const float* b_psb = (const float*)d_in[16];
    const float* W_hsb = (const float*)d_in[17];
    const float* b_hsb = (const float*)d_in[18];
    const float* W_ind = (const float*)d_in[19];
    const float* b_ind = (const float*)d_in[20];
    const float* W_out = (const float*)d_in[21];
    const float* b_out = (const float*)d_in[22];

    const int Nn = N_STOCKS, Cc = N_CONCEPTS;
    char* base = (char*)d_ws;
    size_t off = 0;
    auto alloc = [&](size_t bytes) -> void* {
        off = (off + 255) & ~(size_t)255;
        void* p = base + off;
        off += bytes;
        return p;
    };
    float* x0     = (float*)alloc((size_t)Nn * HID * 4);
    us*    x0n    = (us*)   alloc((size_t)Nn * HID * 2);
    us*    x0r    = (us*)   alloc((size_t)Nn * HID * 2);
    float* colsum = (float*)alloc((size_t)Cc * 4);
    float* irep   = (float*)alloc((size_t)Cc * HID * 4);
    us*    irepn  = (us*)   alloc((size_t)Cc * HID * 2);
    float* pooled = (float*)alloc((size_t)Cc * HID * 4);
    float* urep   = (float*)alloc((size_t)Cc * HID * 4);
    us*    urepn  = (us*)   alloc((size_t)Cc * HID * 2);
    us*    urepr  = (us*)   alloc((size_t)Cc * HID * 2);
    float* attnO  = (float*)alloc((size_t)Nn * HID * 4);
    float* sharedA= (float*)alloc((size_t)Nn * HID * 4);
    float* x1     = (float*)alloc((size_t)Nn * HID * 4);
    us*    x1n    = (us*)   alloc((size_t)Nn * HID * 2);
    float* pfore  = (float*)alloc((size_t)Nn * HID * 4);
    float* diag1  = (float*)alloc((size_t)Nn * 4);
    int*   flags  = (int*)  alloc((size_t)Nn * 4);
    float* topv   = (float*)alloc((size_t)Nn * 3 * 4);
    int*   topi   = (int*)  alloc((size_t)Nn * 3 * 4);
    float* hrep   = (float*)alloc((size_t)Nn * HID * 4);
    us*    hrepn  = (us*)   alloc((size_t)Nn * HID * 2);
    us*    hrepr  = (us*)   alloc((size_t)Nn * HID * 2);
    float* sharedH= (float*)alloc((size_t)Nn * HID * 4);
    float* x2     = (float*)alloc((size_t)Nn * HID * 4);
    float* hfore  = (float*)alloc((size_t)Nn * HID * 4);
    float* indb   = (float*)alloc((size_t)Nn * HID * 4);
    us* bwih0 = (us*)alloc((size_t)384 * 32 * 2);     // padded K: 6 -> 32
    us* bwhh0 = (us*)alloc((size_t)384 * HID * 2);
    us* bwih1 = (us*)alloc((size_t)384 * HID * 2);
    us* bwhh1 = (us*)alloc((size_t)384 * HID * 2);
    us* bWps  = (us*)alloc((size_t)HID * HID * 2);
    us* bWpsf = (us*)alloc((size_t)HID * HID * 2);
    us* bWpsb = (us*)alloc((size_t)HID * HID * 2);
    us* bWhsb = (us*)alloc((size_t)HID * HID * 2);
    us* bWind = (us*)alloc((size_t)HID * HID * 2);

    const int GRU_SH  = 64 * 128 * 2 * 2 + 64 * 32 * 2 + (64 * 256 + 64 * 128 * 2) * 4; // 167936
    const int ATTN_SH = 128 * 128 * (2 * 4) + 128 * 128 * 4 + 3 * 128 * 4;              // 198144
    const int FC_SH   = 2 * 128 * 128 * 2;                                              // 65536
    const int TOPK_SH = 2 * 128 * 128 * 2 + 128 * 128 * 4;                              // 131072
    const int IREP_SH = 2 * 128 * 32 * 2;                                               // 16384
    hipFuncSetAttribute((const void*)k_gru,  hipFuncAttributeMaxDynamicSharedMemorySize, GRU_SH);
    hipFuncSetAttribute((const void*)k_attn, hipFuncAttributeMaxDynamicSharedMemorySize, ATTN_SH);
    hipFuncSetAttribute((const void*)k_fc,   hipFuncAttributeMaxDynamicSharedMemorySize, FC_SH);
    hipFuncSetAttribute((const void*)k_topk, hipFuncAttributeMaxDynamicSharedMemorySize, TOPK_SH);
    hipFuncSetAttribute((const void*)k_irep, hipFuncAttributeMaxDynamicSharedMemorySize, IREP_SH);

    // weight conversion to bf16 (wih0 zero-padded to K=32 for aligned fragments)
    auto cvt = [&](const float* in, us* out, int n) {
        k_cvt<<<(n + 255) / 256, 256, 0, stream>>>(in, out, n);
    };
    k_cvt_pad<<<(384 * 32 + 255) / 256, 256, 0, stream>>>(wih0f, bwih0, 384, IN_D, 32);
    cvt(whh0f, bwhh0, 384 * HID);
    cvt(wih1f, bwih1, 384 * HID);
    cvt(whh1f, bwhh1, 384 * HID);
    cvt(W_ps,  bWps,  HID * HID);
    cvt(W_psf, bWpsf, HID * HID);
    cvt(W_psb, bWpsb, HID * HID);
    cvt(W_hsb, bWhsb, HID * HID);
    cvt(W_ind, bWind, HID * HID);

    k_colsum<<<Cc / 256, 256, 0, stream>>>(CM, MV, colsum, Nn, Cc);
    k_gru<<<Nn / 64, 256, GRU_SH, stream>>>(X, bwih0, bwhh0, bih0, bhh0,
                                            bwih1, bwhh1, bih1, bhh1, x0);
    k_rownorm<<<Nn / 8, 256, 0, stream>>>(x0, Nn, x0n, x0r, nullptr);
    k_irep<<<Cc / 128, 256, IREP_SH, stream>>>(CM, MV, colsum, x0r, irep);
    k_rownorm<<<Cc / 8, 256, 0, stream>>>(irep, Cc, irepn, nullptr, nullptr);
    // s2c_upd^T @ x0  (softmax over stocks): Q=concept reps, K/V=stocks
    k_attn<<<Cc / 128, 256, ATTN_SH, stream>>>(irepn, x0n, x0r, pooled, Nn);
    k_fc<<<Cc / 128, 256, FC_SH, stream>>>(pooled, bWps, b_ps, urep, nullptr, 0);
    k_rownorm<<<Cc / 8, 256, 0, stream>>>(urep, Cc, urepn, urepr, nullptr);
    // c2s @ update_rep (softmax over concepts): Q=stocks, K/V=concepts
    k_attn<<<Nn / 128, 256, ATTN_SH, stream>>>(x0n, urepn, urepr, attnO, Cc);
    k_fc<<<Nn / 128, 256, FC_SH, stream>>>(attnO, bWps, b_ps, sharedA, nullptr, 0);
    k_fc<<<Nn / 128, 256, FC_SH, stream>>>(sharedA, bWpsb, b_psb, x1, x0, 1);   // x1 = x0 - p_back
    k_fc<<<Nn / 128, 256, FC_SH, stream>>>(sharedA, bWpsf, b_psf, pfore, nullptr, 1);
    k_rownorm<<<Nn / 8, 256, 0, stream>>>(x1, Nn, x1n, nullptr, diag1);
    hipMemsetAsync(hrep, 0, (size_t)Nn * HID * 4, stream);
    hipMemsetAsync(flags, 0, (size_t)Nn * 4, stream);
    k_topk<<<Nn / 128, 256, TOPK_SH, stream>>>(x1n, topv, topi);
    k_scatter<<<(Nn * 3) / 8, 256, 0, stream>>>(topv, topi, x1, hrep, flags, Nn * 3);
    k_diag<<<(Nn * HID) / 256, 256, 0, stream>>>(flags, diag1, x1, hrep, Nn * HID);
    k_rownorm<<<Nn / 8, 256, 0, stream>>>(hrep, Nn, hrepn, hrepr, nullptr);
    // c2s_h @ hidden_rep: Q=x1, K/V=hidden reps
    k_attn<<<Nn / 128, 256, ATTN_SH, stream>>>(x1n, hrepn, hrepr, attnO, Nn);
    k_fc<<<Nn / 128, 256, FC_SH, stream>>>(attnO, bWps, b_ps, sharedH, nullptr, 0);
    k_fc<<<Nn / 128, 256, FC_SH, stream>>>(sharedH, bWhsb, b_hsb, x2, x1, 1);   // x2 = x1 - h_back
    k_fc<<<Nn / 128, 256, FC_SH, stream>>>(sharedH, bWpsf, b_psf, hfore, nullptr, 1);
    k_fc<<<Nn / 128, 256, FC_SH, stream>>>(x2, bWind, b_ind, indb, nullptr, 1);
    k_final<<<Nn / 8, 256, 0, stream>>>(pfore, hfore, indb, W_out, b_out, (float*)d_out, Nn);
}
